// RealityStoneCompressedMLP_43069932044594
// MI455X (gfx1250) — matrix operand order repaired and checked by reading, compiled, not run on hardware
//
#include <hip/hip_runtime.h>
#include <hip/hip_bf16.h>
#include <utility>

// ---------------- problem constants ----------------
constexpr int D_MODEL = 1024;
constexpr int D_FF    = 4096;
constexpr int MROWS   = 8192;              // B*S = 4*2048
constexpr int NFFT    = 1 << 22;           // PAD_N
constexpr int MH      = 1 << 21;           // half-size complex IFFT length
constexpr int FFT_LEN = MH + 1;            // rfft bins
constexpr int KEEP    = 524288;

typedef __attribute__((ext_vector_type(16))) _Float16 v16h;
typedef __attribute__((ext_vector_type(8)))  _Float16 v8h;
typedef __attribute__((ext_vector_type(4)))  _Float16 v4h;
typedef __attribute__((ext_vector_type(8)))  float    v8f;

union U16 { v16h v; v8h h[2]; };

// gfx1250 async global->LDS copy path (guarded: falls back if builtins absent)
#if defined(__gfx1250__) && __has_builtin(__builtin_amdgcn_global_load_async_to_lds_b128) && \
    __has_builtin(__builtin_amdgcn_s_wait_asynccnt)
#define USE_ASYNC_LDS 1
#else
#define USE_ASYNC_LDS 0
#endif

#if USE_ASYNC_LDS
typedef int v4i_gcc __attribute__((vector_size(4 * sizeof(int))));
__device__ __forceinline__ void async_copy16(const _Float16* g, _Float16* l) {
    __builtin_amdgcn_global_load_async_to_lds_b128(
        (__attribute__((address_space(1))) v4i_gcc*)(uintptr_t)g,
        (__attribute__((address_space(3))) v4i_gcc*)(uint32_t)(uintptr_t)l,
        0, 0);
}
#endif

// ---------------- FFT pipeline kernels ----------------

__global__ __launch_bounds__(256) void zero_f32(float* __restrict__ p, int n) {
    int i = blockIdx.x * 256 + threadIdx.x;
    if (i < n) p[i] = 0.0f;
}

__global__ __launch_bounds__(256) void scatter_coeffs(float2* __restrict__ X,
                                                      const float* __restrict__ cr,
                                                      const float* __restrict__ ci,
                                                      const int* __restrict__ idx) {
    int i = blockIdx.x * 256 + threadIdx.x;   // i < KEEP (exact grid)
    int k = idx[i];
    X[k] = make_float2(cr[i], ci[i]);
}

// Z[k] = E[k] + i*O'[k],  E = (X[k]+conj(X[M-k]))/2,
// O' = e^{+2pi i k/N} * (X[k]-conj(X[M-k]))/2
__global__ __launch_bounds__(256) void pack_half(float2* __restrict__ Z,
                                                 const float2* __restrict__ X) {
    int k = blockIdx.x * 256 + threadIdx.x;   // k < MH (exact grid)
    float2 A  = X[k];
    float2 Bc = X[MH - k]; Bc.y = -Bc.y;      // conj
    float Er = 0.5f * (A.x + Bc.x), Ei = 0.5f * (A.y + Bc.y);
    float Dr = 0.5f * (A.x - Bc.x), Di = 0.5f * (A.y - Bc.y);
    float th = 6.2831853071795864f * ((float)k / (float)NFFT);
    float c = cosf(th), s = sinf(th);
    float Or = c * Dr - s * Di;
    float Oi = s * Dr + c * Di;
    Z[k] = make_float2(Er - Oi, Ei + Or);     // E + i*O'
}

// radix-2 Stockham autosort pass, inverse (sign = +1), unscaled.
__global__ __launch_bounds__(256) void fft_pass(const float2* __restrict__ src,
                                                float2* __restrict__ dst, int l2Ns) {
    const int half = MH / 2;
    int j = blockIdx.x * 256 + threadIdx.x;   // j < half (exact grid)
    int Ns = 1 << l2Ns;
    float2 a = src[j];
    float2 b = src[j + half];
    int q = j & (Ns - 1);
    float ang = 6.2831853071795864f * ((float)q / (float)(Ns << 1)); // +sign: inverse
    float c = cosf(ang), s = sinf(ang);
    float2 bt = make_float2(c * b.x - s * b.y, s * b.x + c * b.y);
    int d = ((j >> l2Ns) << (l2Ns + 1)) | q;  // (j/Ns)*2Ns + (j%Ns)
    dst[d]      = make_float2(a.x + bt.x, a.y + bt.y);
    dst[d + Ns] = make_float2(a.x - bt.x, a.y - bt.y);
}

// z[n] -> w[2n], w[2n+1]; apply 1/MH (half-size IFFT scale) * user scale; to f16
__global__ __launch_bounds__(256) void unpack_w(_Float16* __restrict__ w,
                                                const float2* __restrict__ z,
                                                const float* __restrict__ scale) {
    int n = blockIdx.x * 256 + threadIdx.x;   // n < MH (exact grid)
    float f = scale[0] * (1.0f / 2097152.0f);
    float2 v = z[n];
    w[2 * n + 0] = (_Float16)(v.x * f);
    w[2 * n + 1] = (_Float16)(v.y * f);
}

// x (f32) -> f16, vectorized by 4
__global__ __launch_bounds__(256) void cvt_f32_f16(const float* __restrict__ in,
                                                   _Float16* __restrict__ out) {
    int i = blockIdx.x * 256 + threadIdx.x;   // i < n/4 (exact grid)
    float4 v = ((const float4*)in)[i];
    v4h o = { (_Float16)v.x, (_Float16)v.y, (_Float16)v.z, (_Float16)v.w };
    ((v4h*)out)[i] = o;
}

// ---------------- WMMA GEMM ----------------
// C[M,N] = A[M,K] * W[K,N] (+bias, optional exact GELU).
// A is f16 row-major. Block: 256 threads (8 wave32), tile 128(M) x 64(N), K-step 32.
// Wave w owns m-block w (16 rows) x 4 n-blocks of 16.
template <bool GELU_F16OUT>
__global__ __launch_bounds__(256) void gemm_wmma(const _Float16* __restrict__ A,
                                                 const _Float16* __restrict__ W,
                                                 const float* __restrict__ bias,
                                                 void* __restrict__ Cp,
                                                 int K, int N) {
    __shared__ alignas(16) _Float16 lA[128][40];   // padded: 80B row stride
    __shared__ alignas(16) _Float16 lB[64][40];    // stored [n][k]

    const int t    = threadIdx.x;
    const int lane = t & 31;
    const int wv   = t >> 5;          // 0..7
    const int m0   = blockIdx.y * 128;
    const int n0   = blockIdx.x * 64;
    const int mr   = lane & 15;
    const int hi   = lane >> 4;       // 0/1

    v8f acc[4] = {};

    for (int k0 = 0; k0 < K; k0 += 32) {
        // ---- stage A tile [128][32] into LDS ----
        {
            int row = t >> 1;
            int cb  = (t & 1) * 16;
            const _Float16* gA = A + (size_t)(m0 + row) * K + k0 + cb;
            _Float16* lp = &lA[row][cb];
#if USE_ASYNC_LDS
            async_copy16(gA, lp);          // global_load_async_to_lds_b128
            async_copy16(gA + 8, lp + 8);
#else
            const v8h* g = (const v8h*)gA;
            *(v8h*)(lp + 0) = g[0];
            *(v8h*)(lp + 8) = g[1];
#endif
        }
        // ---- stage B tile: W[k0..k0+32)[n0..n0+64) -> lB[n][k] (transpose) ----
        {
            int kk = t >> 3;
            int ng = (t & 7) * 8;
            const v8h* g = (const v8h*)(W + (size_t)(k0 + kk) * N + n0 + ng);
            v8h v = *g;
#pragma unroll
            for (int u = 0; u < 8; ++u) lB[ng + u][kk] = v[u];
            if (k0 + 32 < K)   // gfx1250 global_prefetch_b8 of next tile
                __builtin_prefetch((const char*)(W + (size_t)(k0 + 32 + kk) * N + n0 + ng), 0, 1);
        }
#if USE_ASYNC_LDS
        __builtin_amdgcn_s_wait_asynccnt(0);
#endif
        __syncthreads();

        // ---- A fragment: k = (h<8?0:16) + h%8 + (lane>=16?8:0), M = lane%16 ----
        U16 a;
        a.h[0] = *(const v8h*)&lA[wv * 16 + mr][hi * 8];
        a.h[1] = *(const v8h*)&lA[wv * 16 + mr][16 + hi * 8];

        // ---- preload all B fragments, then issue WMMAs back-to-back ----
        U16 bf[4];
#pragma unroll
        for (int j = 0; j < 4; ++j) {
            // B fragment: n = lane%16, k = (h<8?0:16) + h%8 + (lane>=16?8:0)
            bf[j].h[0] = *(const v8h*)&lB[j * 16 + mr][hi * 8];
            bf[j].h[1] = *(const v8h*)&lB[j * 16 + mr][16 + hi * 8];
        }
#pragma unroll
        for (int j = 0; j < 4; ++j) {
            acc[j] = __builtin_amdgcn_wmma_f32_16x16x32_f16(
                false, a.v, false, bf[j].v, (short)0, acc[j], false, false);
        }
        __syncthreads();
    }

    // ---- epilogue: D layout M = r + hi*8, N = lane%16 ----
#pragma unroll
    for (int j = 0; j < 4; ++j) {
        int n = n0 + j * 16 + mr;
        float bs = bias[n];
#pragma unroll
        for (int r = 0; r < 8; ++r) {
            int m = m0 + wv * 16 + r + hi * 8;
            float v = acc[j][r] + bs;
            if (GELU_F16OUT) {
                v = 0.5f * v * (1.0f + erff(v * 0.70710678118f));
                ((_Float16*)Cp)[(size_t)m * N + n] = (_Float16)v;
            } else {
                ((float*)Cp)[(size_t)m * N + n] = v;
            }
        }
    }
}

// ---------------- host-side decompress driver ----------------
static void run_decompress(const float* cr, const float* ci, const int* idx,
                           const float* scale, _Float16* w16,
                           float2* X, float2* ping, float2* pong, hipStream_t s) {
    int nz = 2 * FFT_LEN;
    zero_f32<<<(nz + 255) / 256, 256, 0, s>>>((float*)X, nz);
    scatter_coeffs<<<KEEP / 256, 256, 0, s>>>(X, cr, ci, idx);
    pack_half<<<MH / 256, 256, 0, s>>>(ping, X);
    float2* a = ping; float2* b = pong;
    for (int st = 0; st < 21; ++st) {
        fft_pass<<<(MH / 2) / 256, 256, 0, s>>>(a, b, st);
        std::swap(a, b);
    }
    unpack_w<<<MH / 256, 256, 0, s>>>(w16, a, scale);
}

extern "C" void kernel_launch(void* const* d_in, const int* in_sizes, int n_in,
                              void* d_out, int out_size, void* d_ws, size_t ws_size,
                              hipStream_t stream) {
    const float* x         = (const float*)d_in[0];
    const float* fc_cr     = (const float*)d_in[1];
    const float* fc_ci     = (const float*)d_in[2];
    const int*   fc_idx    = (const int*)  d_in[3];
    const float* fc_scale  = (const float*)d_in[4];
    const float* fc_bias   = (const float*)d_in[5];
    const float* pj_cr     = (const float*)d_in[6];
    const float* pj_ci     = (const float*)d_in[7];
    const int*   pj_idx    = (const int*)  d_in[8];
    const float* pj_scale  = (const float*)d_in[9];
    const float* pj_bias   = (const float*)d_in[10];
    (void)in_sizes; (void)n_in; (void)out_size; (void)ws_size;

    // workspace layout (bytes)
    const size_t MB = 1024ull * 1024ull;
    char* ws = (char*)d_ws;
    float2*    X     = (float2*)(ws);                       // 16MB + 8B (spectrum)
    float2*    ping  = (float2*)(ws + 17 * MB);             // 16MB
    float2*    pong  = (float2*)(ws + 33 * MB);             // 16MB
    _Float16*  Wfc   = (_Float16*)(ws + 49 * MB);           // 8MB  [1024][4096]
    _Float16*  Wpj   = (_Float16*)(ws + 57 * MB);           // 8MB  [4096][1024]
    _Float16*  H     = (_Float16*)(ws + 65 * MB);           // 64MB [8192][4096]
    _Float16*  X16   = (_Float16*)(ws + 129 * MB);          // 16MB [8192][1024] f16 x

    // x -> f16 (independent of decompress, same stream)
    cvt_f32_f16<<<(MROWS * D_MODEL / 4) / 256, 256, 0, stream>>>(x, X16);

    // decompress both weights (stream-ordered, buffers reused)
    run_decompress(fc_cr, fc_ci, fc_idx, fc_scale, Wfc, X, ping, pong, stream);
    run_decompress(pj_cr, pj_ci, pj_idx, pj_scale, Wpj, X, ping, pong, stream);

    // GEMM1: H = gelu(x @ Wfc + b_fc), f16 out
    {
        dim3 grid(D_FF / 64, MROWS / 128);
        gemm_wmma<true><<<grid, 256, 0, stream>>>(X16, Wfc, fc_bias, (void*)H,
                                                  D_MODEL, D_FF);
    }
    // GEMM2: out = H @ Wpj + b_proj, f32 out
    {
        dim3 grid(D_MODEL / 64, MROWS / 128);
        gemm_wmma<false><<<grid, 256, 0, stream>>>(H, Wpj, pj_bias, d_out,
                                                   D_FF, D_MODEL);
    }
}
